// BiMambaAdapter_5755256176889
// MI455X (gfx1250) — matrix operand order repaired and checked by reading, not compiled, not run on hardware
//
#include <hip/hip_runtime.h>
#include <hip/hip_bf16.h>

// ---------------- problem constants ----------------
#define DMODEL 768
#define DINNER 1536
#define DSTATE 16
#define DTRANK 48
#define XDBLW  (DTRANK + 2 * DSTATE)   // 80
#define BSZ    2
#define SEQL   2048
#define MROWS  (BSZ * SEQL)            // 4096

typedef __attribute__((ext_vector_type(16))) __bf16 v16bf;
typedef __attribute__((ext_vector_type(8)))  float  v8f;
typedef unsigned short u16;
typedef unsigned int   u32;
typedef unsigned long long u64;

// ---------------- bf16 helpers ----------------
__device__ __forceinline__ u16 f2bf(float f) {
    u32 u = __float_as_uint(f);
    u32 r = u + 0x7FFFu + ((u >> 16) & 1u);   // round-to-nearest-even
    return (u16)(r >> 16);
}
__device__ __forceinline__ float bf2f(u16 h) {
    return __uint_as_float(((u32)h) << 16);
}

// ---- CDNA5 async global->LDS copy (32 B per lane: two b128 chunks) ----
// INST_OFFSET is added to BOTH the LDS and the global address (ISA 15.18.3),
// so chunk 2 reuses the same address registers with offset:16.
__device__ __forceinline__ void async_cp32(u32 lds_addr, u64 gaddr) {
    asm volatile("global_load_async_to_lds_b128 %0, %1, off\n\t"
                 "global_load_async_to_lds_b128 %0, %1, off offset:16"
                 :: "v"(lds_addr), "v"(gaddr) : "memory");
}
__device__ __forceinline__ void wait_async0() {
    asm volatile("s_wait_asynccnt 0x0" ::: "memory");
}

// ---------------- generic f32 -> bf16 convert ----------------
__global__ void convert_f32_bf16(const float* __restrict__ src, u16* __restrict__ dst, int n) {
    int i = blockIdx.x * 256 + threadIdx.x;
    if (i < n) dst[i] = f2bf(src[i]);
}

// ---------------- LayerNorm -> bf16 (normal + time-reversed copies) ----------------
__global__ void ln_kernel(const float* __restrict__ x, const float* __restrict__ w,
                          const float* __restrict__ b, u16* __restrict__ h,
                          u16* __restrict__ hrev) {
    const int row = blockIdx.x;          // 0 .. MROWS-1
    const int tid = threadIdx.x;         // 256 threads, 768 cols -> 3 each
    __shared__ float red[256];

    float s = 0.f;
    for (int i = 0; i < 3; ++i) s += x[(size_t)row * DMODEL + tid + 256 * i];
    red[tid] = s; __syncthreads();
    for (int off = 128; off > 0; off >>= 1) {
        if (tid < off) red[tid] += red[tid + off];
        __syncthreads();
    }
    const float mu = red[0] / (float)DMODEL;
    __syncthreads();

    float v = 0.f;
    for (int i = 0; i < 3; ++i) {
        float d = x[(size_t)row * DMODEL + tid + 256 * i] - mu;
        v += d * d;
    }
    red[tid] = v; __syncthreads();
    for (int off = 128; off > 0; off >>= 1) {
        if (tid < off) red[tid] += red[tid + off];
        __syncthreads();
    }
    const float rstd = rsqrtf(red[0] / (float)DMODEL + 1e-5f);

    const int bb = row / SEQL, tt = row % SEQL;
    const int rrow = bb * SEQL + (SEQL - 1 - tt);
    for (int i = 0; i < 3; ++i) {
        int c = tid + 256 * i;
        float val = (x[(size_t)row * DMODEL + c] - mu) * rstd * w[c] + b[c];
        u16 bits = f2bf(val);
        h[(size_t)row * DMODEL + c]     = bits;
        hrev[(size_t)rrow * DMODEL + c] = bits;
    }
}

// ---------------- WMMA GEMM:  C[M,N] = A[M,K] * W[N,K]^T ----------------
// block = 256 threads = 8 waves; tile 128(M) x 128(N); K stepped by 32.
// Double-buffered LDS, filled by GLOBAL_LOAD_ASYNC_TO_LDS_B128.
// EPI: 0 = f32 store
//      1 = f32 softplus(acc + bias[n])
//      2 = bf16 store, optional per-batch row reversal (rev_L) + col offset
//      3 = f32 acc + resid[m,n] + bias[n]
#define LDS_STR 40   // padded row stride (halves); keeps b128 LDS ops 16B-aligned

template <int EPI>
__global__ __launch_bounds__(256)
void gemm_wmma(const u16* __restrict__ A, const u16* __restrict__ W,
               int M, int N, int K, int lda, int ldw,
               float* __restrict__ Cf, u16* __restrict__ Cb,
               int ldc, int c_col_off,
               const float* __restrict__ bias,
               const float* __restrict__ resid, int rev_L) {
    __shared__ u16 As[2][128 * LDS_STR];
    __shared__ u16 Ws[2][128 * LDS_STR];

    const int tid  = threadIdx.x;
    const int lane = tid & 31;
    const int wv   = tid >> 5;       // wave id 0..7 -> 16-row group
    const int hl   = lane >> 4;      // lane half
    const int r    = lane & 15;
    const int gm0  = blockIdx.y * 128;
    const int gn0  = blockIdx.x * 128;

    v8f acc[8];
    const v8f vzero = {0.f, 0.f, 0.f, 0.f, 0.f, 0.f, 0.f, 0.f};
#pragma unroll
    for (int f = 0; f < 8; ++f) acc[f] = vzero;

    const int nk = (K + 31) / 32;

    // stage one 128x32 tile of A and W into LDS buffer `buf`
    auto stage = [&](int k0, int buf) {
        const int row  = tid >> 1;
        const int koff = (tid & 1) * 16;
        {   // A tile
            const int grow = gm0 + row;
            if (grow < M && (k0 + koff + 16) <= K) {
                u32 l = (u32)(uintptr_t)(&As[buf][row * LDS_STR + koff]);
                u64 g = (u64)(uintptr_t)(A + (size_t)grow * lda + k0 + koff);
                async_cp32(l, g);
            } else {
                for (int j = 0; j < 16; ++j) {
                    int kg = k0 + koff + j;
                    As[buf][row * LDS_STR + koff + j] =
                        (grow < M && kg < K) ? A[(size_t)grow * lda + kg] : (u16)0;
                }
            }
        }
        {   // W tile
            const int grow = gn0 + row;
            if (grow < N && (k0 + koff + 16) <= K) {
                u32 l = (u32)(uintptr_t)(&Ws[buf][row * LDS_STR + koff]);
                u64 g = (u64)(uintptr_t)(W + (size_t)grow * ldw + k0 + koff);
                async_cp32(l, g);
            } else {
                for (int j = 0; j < 16; ++j) {
                    int kg = k0 + koff + j;
                    Ws[buf][row * LDS_STR + koff + j] =
                        (grow < N && kg < K) ? W[(size_t)grow * ldw + kg] : (u16)0;
                }
            }
        }
    };

    stage(0, 0);
    wait_async0();
    __syncthreads();

    for (int kk = 0; kk < nk; ++kk) {
        const int cur = kk & 1;
        if (kk + 1 < nk) stage((kk + 1) * 32, cur ^ 1);   // overlap with compute

        union Frag { v16bf v; u32 u[8]; };
        Frag af;
#pragma unroll
        for (int v = 0; v < 8; ++v) {
            int kb = ((v < 4) ? (v * 2) : (16 + (v - 4) * 2)) + hl * 8;
            af.u[v] = *(const u32*)(&As[cur][(wv * 16 + r) * LDS_STR + kb]);
        }
#pragma unroll
        for (int f = 0; f < 8; ++f) {
            Frag bfg;
#pragma unroll
            for (int v = 0; v < 8; ++v) {
                int kb = ((v < 4) ? (v * 2) : (16 + (v - 4) * 2)) + hl * 8;
                bfg.u[v] = *(const u32*)(&Ws[cur][(f * 16 + r) * LDS_STR + kb]);
            }
            acc[f] = __builtin_amdgcn_wmma_f32_16x16x32_bf16(
                false, af.v, false, bfg.v, (short)0, acc[f], false, false);
        }
        wait_async0();       // next tile's async copies must have landed
        __syncthreads();     // and everyone is done reading before reuse
    }

    // ---- epilogue (C layout: VGPR v -> M = v + 8*half, N = lane%16)
#pragma unroll
    for (int f = 0; f < 8; ++f) {
#pragma unroll
        for (int v = 0; v < 8; ++v) {
            int m = gm0 + wv * 16 + hl * 8 + v;
            int n = gn0 + f * 16 + r;
            if (m < M && n < N) {
                float val = acc[f][v];
                if (EPI == 0) {
                    Cf[(size_t)m * ldc + n] = val;
                } else if (EPI == 1) {
                    val += bias[n];
                    val = (val > 20.f) ? val : log1pf(__expf(val));   // softplus
                    Cf[(size_t)m * ldc + n] = val;
                } else if (EPI == 2) {
                    int mm = m;
                    if (rev_L > 0) {
                        int b = m / rev_L, t = m % rev_L;
                        mm = b * rev_L + (rev_L - 1 - t);
                    }
                    Cb[(size_t)mm * ldc + c_col_off + n] = f2bf(val);
                } else { // EPI == 3
                    Cf[(size_t)m * ldc + n] = val + resid[(size_t)m * ldc + n] + bias[n];
                }
            }
        }
    }
}

// ---------------- causal depthwise conv (width 4) + bias + SiLU ----------------
__global__ void conv_silu_kernel(const float* __restrict__ xz, const float* __restrict__ cw,
                                 const float* __restrict__ cb, float* __restrict__ uf,
                                 u16* __restrict__ ub) {
    const int idx = blockIdx.x * 256 + threadIdx.x;   // MROWS*DINNER threads
    const int d = idx % DINNER;
    const int m = idx / DINNER;
    const int t = m % SEQL, b = m / SEQL;
    float s = cb[d];
#pragma unroll
    for (int j = 0; j < 4; ++j) {
        int tj = t + j - 3;
        if (tj >= 0)
            s += xz[((size_t)(b * SEQL + tj)) * (2 * DINNER) + d] * cw[d * 4 + j];
    }
    s = s / (1.f + __expf(-s));   // SiLU
    uf[idx] = s;
    ub[idx] = f2bf(s);
}

// ---------------- selective scan (sequential in t, parallel over b,d) ----------------
__global__ void scan_kernel(const float* __restrict__ uf, const float* __restrict__ dtf,
                            const u16* __restrict__ xdbl, const float* __restrict__ xz,
                            const float* __restrict__ A_log, const float* __restrict__ Dp,
                            u16* __restrict__ y) {
    const int d = (blockIdx.x % (DINNER / 256)) * 256 + threadIdx.x;
    const int b = blockIdx.x / (DINNER / 256);

    float Ad[DSTATE];
#pragma unroll
    for (int n = 0; n < DSTATE; ++n) Ad[n] = -__expf(A_log[d * DSTATE + n]);
    const float Dd = Dp[d];
    float hs[DSTATE];
#pragma unroll
    for (int n = 0; n < DSTATE; ++n) hs[n] = 0.f;

    __shared__ float bc[8][32];   // [step][0..15]=B, [16..31]=C
    for (int t0 = 0; t0 < SEQL; t0 += 8) {
        {
            int tt = threadIdx.x >> 5, i = threadIdx.x & 31;
            bc[tt][i] = bf2f(xdbl[((size_t)(b * SEQL + t0 + tt)) * XDBLW + DTRANK + i]);
        }
        __syncthreads();
#pragma unroll
        for (int s = 0; s < 8; ++s) {
            const int t = t0 + s;
            const size_t base = (size_t)(b * SEQL + t);
            const float u   = uf[base * DINNER + d];
            const float dtv = dtf[base * DINNER + d];
            const float du  = dtv * u;
            float yv = 0.f;
#pragma unroll
            for (int n = 0; n < DSTATE; ++n) {
                float dA = __expf(dtv * Ad[n]);
                hs[n] = hs[n] * dA + du * bc[s][n];
                yv += hs[n] * bc[s][16 + n];
            }
            yv += u * Dd;
            float z = xz[base * (2 * DINNER) + DINNER + d];
            yv *= z / (1.f + __expf(-z));   // gate with silu(z)
            y[base * DINNER + d] = f2bf(yv);
        }
        __syncthreads();
    }
}

// ---------------- host-side orchestration ----------------
static inline void* carve(char*& p, size_t bytes) {
    void* r = (void*)p;
    p += (bytes + 255) & ~(size_t)255;
    return r;
}

extern "C" void kernel_launch(void* const* d_in, const int* in_sizes, int n_in,
                              void* d_out, int out_size, void* d_ws, size_t ws_size,
                              hipStream_t stream) {
    (void)in_sizes; (void)n_in; (void)out_size; (void)ws_size;

    const float* x      = (const float*)d_in[0];
    const float* ln_w   = (const float*)d_in[1];
    const float* ln_b   = (const float*)d_in[2];
    const float* in_w[2]    = {(const float*)d_in[3],  (const float*)d_in[12]};
    const float* conv_w[2]  = {(const float*)d_in[4],  (const float*)d_in[13]};
    const float* conv_b[2]  = {(const float*)d_in[5],  (const float*)d_in[14]};
    const float* xproj_w[2] = {(const float*)d_in[6],  (const float*)d_in[15]};
    const float* dt_w[2]    = {(const float*)d_in[7],  (const float*)d_in[16]};
    const float* dt_b[2]    = {(const float*)d_in[8],  (const float*)d_in[17]};
    const float* A_log[2]   = {(const float*)d_in[9],  (const float*)d_in[18]};
    const float* Dp[2]      = {(const float*)d_in[10], (const float*)d_in[19]};
    const float* out_w[2]   = {(const float*)d_in[11], (const float*)d_in[20]};
    const float* proj_w = (const float*)d_in[21];
    const float* proj_b = (const float*)d_in[22];
    float* out = (float*)d_out;

    // ---- workspace carve
    char* p = (char*)d_ws;
    u16* h_bf    = (u16*)carve(p, (size_t)MROWS * DMODEL * 2);
    u16* hrev_bf = (u16*)carve(p, (size_t)MROWS * DMODEL * 2);
    u16* inw_bf[2]; u16* xpw_bf[2]; u16* dtw_bf[2]; u16* outw_bf[2];
    for (int d = 0; d < 2; ++d) {
        inw_bf[d]  = (u16*)carve(p, (size_t)2 * DINNER * DMODEL * 2);
        xpw_bf[d]  = (u16*)carve(p, (size_t)XDBLW * DINNER * 2);
        dtw_bf[d]  = (u16*)carve(p, (size_t)DINNER * DTRANK * 2);
        outw_bf[d] = (u16*)carve(p, (size_t)DMODEL * DINNER * 2);
    }
    u16* projw_bf = (u16*)carve(p, (size_t)DMODEL * DINNER * 2);
    float* xz[2]; float* uf[2]; u16* ub[2]; u16* xdbl[2]; float* dtf[2]; u16* yb[2];
    for (int d = 0; d < 2; ++d) {
        xz[d]   = (float*)carve(p, (size_t)MROWS * 2 * DINNER * 4);
        uf[d]   = (float*)carve(p, (size_t)MROWS * DINNER * 4);
        ub[d]   = (u16*)carve(p, (size_t)MROWS * DINNER * 2);
        xdbl[d] = (u16*)carve(p, (size_t)MROWS * XDBLW * 2);
        dtf[d]  = (float*)carve(p, (size_t)MROWS * DINNER * 4);
        yb[d]   = (u16*)carve(p, (size_t)MROWS * DINNER * 2);
    }
    u16* comb = (u16*)carve(p, (size_t)MROWS * DINNER * 2);

    // ---- weight conversions
    auto cvt = [&](const float* s, u16* d, int n) {
        convert_f32_bf16<<<dim3((n + 255) / 256), dim3(256), 0, stream>>>(s, d, n);
    };
    for (int d = 0; d < 2; ++d) {
        cvt(in_w[d],    inw_bf[d],  2 * DINNER * DMODEL);
        cvt(xproj_w[d], xpw_bf[d],  XDBLW * DINNER);
        cvt(dt_w[d],    dtw_bf[d],  DINNER * DTRANK);
        cvt(out_w[d],   outw_bf[d], DMODEL * DINNER);
    }
    cvt(proj_w, projw_bf, DMODEL * DINNER);

    // ---- layernorm (+ reversed copy for the backward direction)
    ln_kernel<<<dim3(MROWS), dim3(256), 0, stream>>>(x, ln_w, ln_b, h_bf, hrev_bf);

    const dim3 blk(256);
    auto ggrid = [](int N) { return dim3((N + 127) / 128, MROWS / 128); };

    for (int d = 0; d < 2; ++d) {
        const u16* A = (d == 0) ? h_bf : hrev_bf;
        // in-projection: xz = h @ in_w^T   (M=4096, K=768, N=3072)
        gemm_wmma<0><<<ggrid(2 * DINNER), blk, 0, stream>>>(
            A, inw_bf[d], MROWS, 2 * DINNER, DMODEL, DMODEL, DMODEL,
            xz[d], nullptr, 2 * DINNER, 0, nullptr, nullptr, 0);
        // depthwise conv + silu
        conv_silu_kernel<<<dim3((MROWS * DINNER) / 256), blk, 0, stream>>>(
            xz[d], conv_w[d], conv_b[d], uf[d], ub[d]);
        // x-projection: x_dbl = u @ xproj_w^T   (K=1536, N=80) -> bf16
        gemm_wmma<2><<<ggrid(XDBLW), blk, 0, stream>>>(
            ub[d], xpw_bf[d], MROWS, XDBLW, DINNER, DINNER, DINNER,
            nullptr, xdbl[d], XDBLW, 0, nullptr, nullptr, 0);
        // dt-projection + softplus: dt = softplus(dt_in @ dt_w^T + dt_b)  (K=48, N=1536)
        gemm_wmma<1><<<ggrid(DINNER), blk, 0, stream>>>(
            xdbl[d], dtw_bf[d], MROWS, DINNER, DTRANK, XDBLW, DTRANK,
            dtf[d], nullptr, DINNER, 0, dt_b[d], nullptr, 0);
        // selective scan + D-skip + z-gate
        scan_kernel<<<dim3(BSZ * (DINNER / 256)), blk, 0, stream>>>(
            uf[d], dtf[d], xdbl[d], xz[d], A_log[d], Dp[d], yb[d]);
        // out-projection into concat buffer (backward dir re-reversed in epilogue)
        gemm_wmma<2><<<ggrid(DMODEL), blk, 0, stream>>>(
            yb[d], outw_bf[d], MROWS, DMODEL, DINNER, DINNER, DINNER,
            nullptr, comb, DINNER, d * DMODEL, nullptr, nullptr,
            (d == 0) ? 0 : SEQL);
    }

    // final projection: out = x + combined @ proj_w^T + proj_b  (K=1536, N=768)
    gemm_wmma<3><<<ggrid(DMODEL), blk, 0, stream>>>(
        comb, projw_bf, MROWS, DMODEL, DINNER, DINNER, DINNER,
        out, nullptr, DMODEL, 0, proj_b, x, 0);
}
